// VQ_VAE_6554120093899
// MI455X (gfx1250) — matrix operand order repaired and checked by reading, not compiled, not run on hardware
//
#include <hip/hip_runtime.h>
#include <hip/hip_bf16.h>
#include <math.h>

typedef __attribute__((ext_vector_type(16))) __bf16 v16bf;
typedef __attribute__((ext_vector_type(8)))  __bf16 v8bf;
typedef __attribute__((ext_vector_type(8)))  float  v8f;

static __device__ __forceinline__ __bf16 f2bf(float f) {
    unsigned int u = __builtin_bit_cast(unsigned int, f);
    u += 0x7fffu + ((u >> 16) & 1u);                 // round-to-nearest-even
    unsigned short s = (unsigned short)(u >> 16);
    return __builtin_bit_cast(__bf16, s);
}
static __device__ __forceinline__ float bf2f(__bf16 h) {
    unsigned short s = __builtin_bit_cast(unsigned short, h);
    unsigned int u = ((unsigned int)s) << 16;
    return __builtin_bit_cast(float, u);
}

// ---------------------------------------------------------------------------
// Implicit-GEMM conv / conv-transpose on bf16 NHWC activations.
//   D[M=pixels][N=cout] = A[M][K] * B[K][N],  K = KS*KS*Cin (tap-major, ci-minor)
// Block = 256 threads = 8 waves; wave w owns M-tile (blockIdx.x*8+w); all waves
// share 4 16-wide N-tiles. The 32k x 64n B panel is staged into LDS each K-step
// with gfx1250 GLOBAL_LOAD_ASYNC_TO_LDS_B128 (ASYNCcnt-tracked, no VGPR round
// trip), guard-free (weights pre-padded to [Coutp%64][Kp%32]).
// A loads are unconditional b128s: OOB/padding chunks are redirected to a
// zero page via address cndmask (no exec-mask divergence on the load itself).
// TRANS=1: ConvTranspose2d(stride2, pad1, opad1), flip folded into packed W.
// smode: 0 = bf16 NHWC out (+bias), 1 = f32 [P][Cout] out, 2 = f32 NCHW tanh.
// Requires: Cin = 1<<log2Cin (>=8), P % 128 == 0, H/W powers of two.
// ---------------------------------------------------------------------------
template<int KS, int STRIDE, int PAD, int TRANS>
__global__ void __launch_bounds__(256)
conv_fast(const __bf16* __restrict__ act, const __bf16* __restrict__ wp,
          const float* __restrict__ bias, void* __restrict__ outp,
          const __bf16* __restrict__ zpage,
          int log2Cin, int Hin, int Win, int Cout, int K, int Kp,
          int log2Hout, int log2Wout, int smode)
{
    constexpr int NT = 4;
    const int tid  = threadIdx.x;
    const int wave = tid >> 5;
    const int lane = tid & 31;
    const int kh   = lane >> 4;           // K-half selector
    const int row  = lane & 15;           // A-row / D-column within tile
    const int tileM   = blockIdx.x * 8 + wave;
    const int colbase = blockIdx.y * 64;
    const int Wout = 1 << log2Wout;
    const int Hout = 1 << log2Hout;

    // pixel handled by this lane's A rows (both halves cover the same 16 rows)
    const int m  = tileM * 16 + row;
    const int ox = m & (Wout - 1);
    const int oy = (m >> log2Wout) & (Hout - 1);
    const int nb = m >> (log2Wout + log2Hout);

    __shared__ __bf16 ldsB[64 * 40];      // [n_local][k 0..31 + 8 pad], rows 80B

    const int nl   = tid >> 2;            // 0..63 : B row this thread stages
    const int part = tid & 3;             // which 8-chunk of the 32-k row

    v8f acc[NT];
#pragma unroll
    for (int t = 0; t < NT; ++t)
#pragma unroll
        for (int i = 0; i < 8; ++i) acc[t][i] = 0.0f;

    // base for the guard-free KS==1 path (pixel identity mapping)
    const __bf16* actm = act + ((size_t)m << log2Cin);

    auto loadA = [&](int k0) -> v8bf {    // one 8-wide bf16 chunk of A row m
        if (KS == 1) {                    // dense 1x1: always in-bounds
            return *(const v8bf*)(actm + k0);
        }
        const int tap = k0 >> log2Cin;    // may exceed KS*KS in padded tail
        const int ci0 = k0 & ((1 << log2Cin) - 1);
        const int ky  = tap / KS;
        const int kx  = tap - ky * KS;
        int iy, ix; bool ok = (k0 < K);
        if (!TRANS) {
            iy = oy * STRIDE - PAD + ky;
            ix = ox * STRIDE - PAD + kx;
            ok = ok && ((unsigned)iy < (unsigned)Hin) && ((unsigned)ix < (unsigned)Win);
        } else {                          // dilated gather, output stride 2
            const int ty = oy + ky - (KS - 1 - PAD);
            const int tx = ox + kx - (KS - 1 - PAD);
            ok = ok && (ty >= 0) && (tx >= 0) && !(ty & 1) && !(tx & 1);
            iy = ty >> 1; ix = tx >> 1;
            ok = ok && (iy < Hin) && (ix < Win);
        }
        const size_t off = (((size_t)(nb * Hin + iy) * Win + ix) << log2Cin) + ci0;
        const __bf16* p = ok ? (act + off) : zpage;   // address select, no branch
        return *(const v8bf*)p;
    };

    // invariant addresses for this thread's async B-stage chunk:
    // LDS byte offset within the (sole) group-segment allocation
    const unsigned ldsoff = (unsigned)(nl * 80 + part * 16);
    const __bf16* wrow = wp + (size_t)(colbase + nl) * Kp + part * 8;

    for (int kk = 0; kk < Kp; kk += 32) {
        // async stage of B panel (64 rows x 32 k), direct global->LDS, no guards
        {
            const unsigned long long ga = (unsigned long long)(size_t)(wrow + kk);
            asm volatile("global_load_async_to_lds_b128 %0, %1, off"
                         :: "v"(ldsoff), "v"(ga) : "memory");
        }

        const v8bf a0 = loadA(kk + 8 * kh);
        const v8bf a1 = loadA(kk + 16 + 8 * kh);
        const v16bf af = __builtin_shufflevector(a0, a1,
            0, 1, 2, 3, 4, 5, 6, 7, 8, 9, 10, 11, 12, 13, 14, 15);

        asm volatile("s_wait_asynccnt 0" ::: "memory");
        __syncthreads();

#pragma unroll
        for (int nt = 0; nt < NT; ++nt) {
            const __bf16* bp = ldsB + (nt * 16 + row) * 40 + kh * 8;
            const v8bf b0 = *(const v8bf*)(bp);
            const v8bf b1 = *(const v8bf*)(bp + 16);
            const v16bf bfv = __builtin_shufflevector(b0, b1,
                0, 1, 2, 3, 4, 5, 6, 7, 8, 9, 10, 11, 12, 13, 14, 15);
            acc[nt] = __builtin_amdgcn_wmma_f32_16x16x32_bf16(
                          false, af, false, bfv, (short)0, acc[nt], false, false);
        }
        __syncthreads();
    }

    // ------------------------- store D ---------------------------------
#pragma unroll
    for (int nt = 0; nt < NT; ++nt) {
        const int co = colbase + nt * 16 + row;
        if (co >= Cout) continue;
        const float bv = bias ? bias[co] : 0.0f;
#pragma unroll
        for (int r = 0; r < 8; ++r) {
            const int mm = tileM * 16 + r + 8 * kh;
            const float v = acc[nt][r] + bv;
            if (smode == 0) {
                ((__bf16*)outp)[(size_t)mm * Cout + co] = f2bf(v);
            } else if (smode == 1) {
                ((float*)outp)[(size_t)mm * Cout + co] = v;
            } else {
                const int oox = mm & (Wout - 1);
                const int ooy = (mm >> log2Wout) & (Hout - 1);
                const int nn  = mm >> (log2Wout + log2Hout);
                const size_t oi =
                    (((((size_t)(nn * Cout + co)) << log2Hout) + ooy) << log2Wout) + oox;
                ((float*)outp)[oi] = tanhf(v);
            }
        }
    }
}

// ---------------------------------------------------------------------------
// Pack f32 weights -> bf16 [Coutp][Kp] (zero-padded), k = tap*Cinp + ci.
// trans=0: src OIHW (Cout,Cin,KS,KS).  trans=1: src (Cin,Cout,KS,KS), flipped.
// ---------------------------------------------------------------------------
__global__ void pack_weights_kernel(const float* __restrict__ w, __bf16* __restrict__ wp,
                                    int Cout, int Coutp, int Cin, int log2Cinp,
                                    int KS, int K, int Kp, int trans)
{
    const int total = Coutp * Kp;
    const int idx = blockIdx.x * blockDim.x + threadIdx.x;
    if (idx >= total) return;
    const int co = idx / Kp;
    const int k  = idx - co * Kp;
    const int tap = k >> log2Cinp;
    const int ci  = k & ((1 << log2Cinp) - 1);
    float val = 0.0f;
    if (co < Cout && k < K && ci < Cin) {
        const int ky = tap / KS;
        const int kx = tap - ky * KS;
        if (!trans)
            val = w[((size_t)(co * Cin + ci) * KS + ky) * KS + kx];
        else
            val = w[((size_t)(ci * Cout + co) * KS + (KS - 1 - ky)) * KS + (KS - 1 - kx)];
    }
    wp[idx] = f2bf(val);
}

// input x: f32 NCHW (64,3,128,128) -> bf16 NHWC (pixels, 8) channel-padded
__global__ void pack_input_kernel(const float* __restrict__ x, __bf16* __restrict__ xp,
                                  int total)
{
    const int i = blockIdx.x * blockDim.x + threadIdx.x;
    if (i >= total) return;
    const int pix = i >> 3;
    const int c   = i & 7;
    const int n   = pix >> 14;            // 128*128 pixels per image
    const int r   = pix & 16383;
    float val = 0.0f;
    if (c < 3) val = x[(((size_t)n * 3 + c) << 14) + r];
    xp[i] = f2bf(val);
}

// BatchNorm batch stats on bf16 NHWC (C = power of two)
__global__ void bn_stats_bf_kernel(const __bf16* __restrict__ x, float* __restrict__ stats,
                                   int P, int C, int log2C)
{
    const int t = blockIdx.x * blockDim.x + threadIdx.x;
    const int c = t & (C - 1);
    const int r0 = t >> log2C;
    const int rstep = (gridDim.x * blockDim.x) >> log2C;
    float s = 0.f, s2 = 0.f;
    for (int r = r0; r < P; r += rstep) {
        const float v = bf2f(x[((size_t)r << log2C) + c]);
        s += v; s2 += v * v;
    }
    atomicAdd(&stats[c], s);
    atomicAdd(&stats[C + c], s2);
}

__global__ void bn_apply_bf_kernel(__bf16* __restrict__ x, const float* __restrict__ stats,
                                   const float* __restrict__ g, const float* __restrict__ b,
                                   size_t total, int C, float invcnt)
{
    for (size_t i = blockIdx.x * (size_t)blockDim.x + threadIdx.x; i < total;
         i += (size_t)gridDim.x * blockDim.x) {
        const int c = (int)(i & (size_t)(C - 1));
        const float mean = stats[c] * invcnt;
        const float var  = stats[C + c] * invcnt - mean * mean;
        const float inv  = rsqrtf(var + 1e-5f);
        float v = (bf2f(x[i]) - mean) * inv * g[c] + b[c];
        x[i] = f2bf(v > 0.f ? v : 0.f);
    }
}

__global__ void cnorm_kernel(const float* __restrict__ cb, float* __restrict__ cn,
                             int Kcodes, int C)
{
    const int j = blockIdx.x * blockDim.x + threadIdx.x;
    if (j >= Kcodes) return;
    float s = 0.f;
    for (int c = 0; c < C; ++c) { const float v = cb[j * C + c]; s += v * v; }
    cn[j] = s;
}

// argmin over 512 codes (contiguous score row per pixel), straight-through
// quantized bf16 output, embed/commit loss accumulation.
__global__ void vq_argmin_kernel(const float* __restrict__ S, const float* __restrict__ cn,
                                 const __bf16* __restrict__ z, const float* __restrict__ cb,
                                 __bf16* __restrict__ q, float* __restrict__ idx_out,
                                 float* __restrict__ lossacc, int P)
{
    const int m = blockIdx.x * blockDim.x + threadIdx.x;
    if (m >= P) return;
    const float* srow = S + (size_t)m * 512;
    int best = 0; float bestd = 3.4e38f;
    for (int j = 0; j < 512; ++j) {
        const float d = cn[j] - 2.0f * srow[j];
        if (d < bestd) { bestd = d; best = j; }
    }
    idx_out[m] = (float)best;
    float ls = 0.f;
    for (int c = 0; c < 64; ++c) {
        const float zc = bf2f(z[(size_t)m * 64 + c]);
        const float qc = cb[best * 64 + c];
        q[(size_t)m * 64 + c] = f2bf(qc);
        const float d = zc - qc;
        ls += d * d;
    }
    atomicAdd(lossacc, ls);
}

__global__ void finalize_loss_kernel(const float* __restrict__ lossacc,
                                     float* __restrict__ out_losses, float scale)
{
    if (blockIdx.x == 0 && threadIdx.x == 0) {
        const float l = lossacc[0] * scale;
        out_losses[0] = l;   // loss_embed (same forward value)
        out_losses[1] = l;   // loss_commit
    }
}

// ---------------------------------------------------------------------------
extern "C" void kernel_launch(void* const* d_in, const int* in_sizes, int n_in,
                              void* d_out, int out_size, void* d_ws, size_t ws_size,
                              hipStream_t stream)
{
    (void)in_sizes; (void)n_in; (void)out_size; (void)ws_size;

    const float* x      = (const float*)d_in[0];
    const float* enc1_w = (const float*)d_in[1];
    const float* enc1_b = (const float*)d_in[2];
    const float* bn1_g  = (const float*)d_in[3];
    const float* bn1_b  = (const float*)d_in[4];
    const float* enc2_w = (const float*)d_in[5];
    const float* enc2_b = (const float*)d_in[6];
    const float* pre_w  = (const float*)d_in[7];
    const float* pre_b  = (const float*)d_in[8];
    const float* cb     = (const float*)d_in[9];
    const float* post_w = (const float*)d_in[10];
    const float* post_b = (const float*)d_in[11];
    const float* dec1_w = (const float*)d_in[12];
    const float* dec1_b = (const float*)d_in[13];
    const float* dbn1_g = (const float*)d_in[14];
    const float* dbn1_b = (const float*)d_in[15];
    const float* dec2_w = (const float*)d_in[16];
    const float* dec2_b = (const float*)d_in[17];

    float* out = (float*)d_out;

    const int B = 64, IMG = 128, H1 = 64, H2 = 32;
    const int P0 = B * IMG * IMG;     // 1,048,576  (input / recon pixels)
    const int P1 = B * H1 * H1;       //   262,144
    const int P2 = B * H2 * H2;       //    65,536
    const int RECON = B * 3 * IMG * IMG;

    // ---------------- workspace (256B-aligned chunks) -------------------
    size_t off = 0;
    char* base = (char*)d_ws;
    auto alloc = [&](size_t nbytes) -> void* {
        void* p = base + off;
        off += (nbytes + 255) & ~(size_t)255;
        return p;
    };
    __bf16* xpack = (__bf16*)alloc((size_t)P0 * 8  * 2);   //  16MB
    __bf16* actA  = (__bf16*)alloc((size_t)P1 * 128 * 2);  //  64MB  h1 / d1
    __bf16* actB  = (__bf16*)alloc((size_t)P2 * 256 * 2);  //  32MB  h2 / dpost
    __bf16* zb    = (__bf16*)alloc((size_t)P2 * 64 * 2);   //   8MB
    __bf16* qb    = (__bf16*)alloc((size_t)P2 * 64 * 2);   //   8MB
    float*  Sb    = (float*) alloc((size_t)P2 * 512 * 4);  // 128MB scores
    __bf16* wp1   = (__bf16*)alloc(128 * 96   * 2);        // padded [Coutp][Kp]
    __bf16* wp2   = (__bf16*)alloc(256 * 1152 * 2);
    __bf16* wp3   = (__bf16*)alloc(64  * 256  * 2);
    __bf16* wpc   = (__bf16*)alloc(512 * 64   * 2);
    __bf16* wp4   = (__bf16*)alloc(256 * 64   * 2);
    __bf16* wp5   = (__bf16*)alloc(128 * 2304 * 2);
    __bf16* wp6   = (__bf16*)alloc(64  * 1152 * 2);
    __bf16* zpage = (__bf16*)alloc(256);                   // zero page for OOB A
    float* stats   = (float*)alloc(2 * 128 * 4);
    float* cnormws = (float*)alloc(512 * 4);
    float* lossacc = (float*)alloc(4);

    hipMemsetAsync(zpage, 0, 256, stream);

    // ---------------- packing passes ------------------------------------
    pack_input_kernel<<<(P0 * 8 + 255) / 256, 256, 0, stream>>>(x, xpack, P0 * 8);
    //                     w       wp   Cout Coutp Cin log2Cinp KS  K    Kp   trans
    pack_weights_kernel<<<(128 * 96   + 255) / 256, 256, 0, stream>>>(enc1_w, wp1, 128, 128, 3,   3, 3, 72,   96,   0);
    pack_weights_kernel<<<(256 * 1152 + 255) / 256, 256, 0, stream>>>(enc2_w, wp2, 256, 256, 128, 7, 3, 1152, 1152, 0);
    pack_weights_kernel<<<(64  * 256  + 255) / 256, 256, 0, stream>>>(pre_w,  wp3, 64,  64,  256, 8, 1, 256,  256,  0);
    pack_weights_kernel<<<(512 * 64   + 255) / 256, 256, 0, stream>>>(cb,     wpc, 512, 512, 64,  6, 1, 64,   64,   0);
    pack_weights_kernel<<<(256 * 64   + 255) / 256, 256, 0, stream>>>(post_w, wp4, 256, 256, 64,  6, 1, 64,   64,   0);
    pack_weights_kernel<<<(128 * 2304 + 255) / 256, 256, 0, stream>>>(dec1_w, wp5, 128, 128, 256, 8, 3, 2304, 2304, 1);
    pack_weights_kernel<<<(64  * 1152 + 255) / 256, 256, 0, stream>>>(dec2_w, wp6, 3,   64,  128, 7, 3, 1152, 1152, 1);

    // ---------------- network -------------------------------------------
    // enc1: 3x3 s2 p1, 8(pad3)->128, 128->64
    conv_fast<3, 2, 1, 0><<<dim3(P1 / 128, 2), 256, 0, stream>>>(
        xpack, wp1, enc1_b, actA, zpage, 3, IMG, IMG, 128, 72, 96, 6, 6, 0);

    hipMemsetAsync(stats, 0, 2 * 128 * sizeof(float), stream);
    bn_stats_bf_kernel<<<256, 256, 0, stream>>>(actA, stats, P1, 128, 7);
    bn_apply_bf_kernel<<<2048, 256, 0, stream>>>(actA, stats, bn1_g, bn1_b,
        (size_t)P1 * 128, 128, 1.0f / (float)P1);

    // enc2: 3x3 s2 p1, 128->256, 64->32
    conv_fast<3, 2, 1, 0><<<dim3(P2 / 128, 4), 256, 0, stream>>>(
        actA, wp2, enc2_b, actB, zpage, 7, H1, H1, 256, 1152, 1152, 5, 5, 0);

    // pre: 1x1, 256->64
    conv_fast<1, 1, 0, 0><<<dim3(P2 / 128, 1), 256, 0, stream>>>(
        actB, wp3, pre_b, zb, zpage, 8, H2, H2, 64, 256, 256, 5, 5, 0);

    // VQ scores S[p][j] = z_p . c_j  (f32 pixel-major for streaming argmin)
    conv_fast<1, 1, 0, 0><<<dim3(P2 / 128, 8), 256, 0, stream>>>(
        zb, wpc, nullptr, Sb, zpage, 6, H2, H2, 512, 64, 64, 5, 5, 1);

    cnorm_kernel<<<2, 256, 0, stream>>>(cb, cnormws, 512, 64);
    hipMemsetAsync(lossacc, 0, sizeof(float), stream);
    vq_argmin_kernel<<<P2 / 256, 256, 0, stream>>>(
        Sb, cnormws, zb, cb, qb, out + RECON + 2, lossacc, P2);

    // post: 1x1, 64->256
    conv_fast<1, 1, 0, 0><<<dim3(P2 / 128, 4), 256, 0, stream>>>(
        qb, wp4, post_b, actB, zpage, 6, H2, H2, 256, 64, 64, 5, 5, 0);

    // dec1: convT 3x3 s2 p1 op1, 256->128, 32->64
    conv_fast<3, 2, 1, 1><<<dim3(P1 / 128, 2), 256, 0, stream>>>(
        actB, wp5, dec1_b, actA, zpage, 8, H2, H2, 128, 2304, 2304, 6, 6, 0);

    hipMemsetAsync(stats, 0, 2 * 128 * sizeof(float), stream);
    bn_stats_bf_kernel<<<256, 256, 0, stream>>>(actA, stats, P1, 128, 7);
    bn_apply_bf_kernel<<<2048, 256, 0, stream>>>(actA, stats, dbn1_g, dbn1_b,
        (size_t)P1 * 128, 128, 1.0f / (float)P1);

    // dec2: convT 3x3 s2 p1 op1, 128->3, 64->128, tanh -> f32 NCHW recon
    conv_fast<3, 2, 1, 1><<<dim3(P0 / 128, 1), 256, 0, stream>>>(
        actA, wp6, dec2_b, out, zpage, 7, H1, H1, 3, 1152, 1152, 7, 7, 2);

    finalize_loss_kernel<<<1, 32, 0, stream>>>(
        lossacc, out + RECON, 1.0f / (float)((size_t)P2 * 64));
}
